// FSSM_Block_24292335026639
// MI455X (gfx1250) — matrix operand order repaired and checked by reading, not compiled, not run on hardware
//
#include <hip/hip_runtime.h>
#include <hip/hip_bf16.h>

// S4/DPLR SSM block for MI455X (gfx1250, wave32).
//  1) cauchy_kernel : atRoots(H,L), one block per channel, coalesced writes
//  2) kgen_kernel   : ifft(4096) -> K -> fft(8192) -> Kf      (LDS FFT, 64KB)
//  3) xpose_kernel  : x (B,L,H) -> xT (B,H,L)                 (LDS 32x32 tiles)
//  4) conv_kernel   : per (b,h): fft(xT row)*Kf -> ifft -> h1T (contiguous)
//  5) packT_kernel  : y1 = gelu(h1 + D*x) -> f16 (M,K), fused transpose
//  6) wconv_kernel  : W_fc -> f16
//  7) init_max      : zero max accumulators (determinism)
//  8) fc_gemm       : W panel staged to LDS via global_load_async_to_lds_b128,
//                     fully-unrolled K-loop, batched B fragments +
//                     double-buffered A, v_wmma_f32_16x16x32_f16
//  9) ln_kernel     : LayerNorm -> d_out, atomic max |y|,|x|
// 10) ratio_kernel  : h_ratio -> d_out tail

#define B_SZ 8
#define L_SEQ 4096
#define H_CH 256
#define N_ST 64
#define NFFT 8192

typedef __attribute__((ext_vector_type(16))) _Float16 v16h;
typedef __attribute__((ext_vector_type(8)))  float    v8f;
typedef int v4i __attribute__((vector_size(16)));

#if defined(__has_builtin)
#if __has_builtin(__builtin_amdgcn_global_load_async_to_lds_b128)
#define HAVE_ASYNC_LDS 1
#endif
#if __has_builtin(__builtin_amdgcn_s_wait_asynccnt)
#define HAVE_WAIT_ASYNC 1
#endif
#endif

#ifdef HAVE_ASYNC_LDS
typedef __attribute__((address_space(1))) v4i global_v4i;   // global (AS1) int4
typedef __attribute__((address_space(3))) v4i lds_v4i;      // LDS (AS3) int4
#endif

__device__ __forceinline__ float2 cmul(float2 a, float2 b) {
    return make_float2(a.x * b.x - a.y * b.y, a.x * b.y + a.y * b.x);
}
__device__ __forceinline__ float2 cadd(float2 a, float2 b) { return make_float2(a.x + b.x, a.y + b.y); }
__device__ __forceinline__ float2 csub(float2 a, float2 b) { return make_float2(a.x - b.x, a.y - b.y); }
__device__ __forceinline__ float2 cinv(float2 a) {
    float d = 1.0f / (a.x * a.x + a.y * a.y);
    return make_float2(a.x * d, -a.y * d);
}
__device__ __forceinline__ float2 cdiv(float2 a, float2 b) { return cmul(a, cinv(b)); }
__device__ __forceinline__ float2 cscale(float2 a, float s) { return make_float2(a.x * s, a.y * s); }

__device__ __forceinline__ float gelu_exact(float v) {
    return 0.5f * v * (1.0f + erff(v * 0.70710678118654752f));
}

#define PI_F 3.14159265358979323846f

// ---------------------------------------------------------------------------
// In-LDS iterative radix-2 FFT. sign=-1 forward, sign=+1 inverse (unscaled).
// ---------------------------------------------------------------------------
__device__ void block_fft(float2* s, int n, int logn, float sign, int tid, int nt) {
    for (int i = tid; i < n; i += nt) {
        int j = (int)(__brev((unsigned)i) >> (32 - logn));
        if (j > i) { float2 t = s[i]; s[i] = s[j]; s[j] = t; }
    }
    __syncthreads();
    for (int st = 1; st <= logn; ++st) {
        int m = 1 << st;
        int mh = m >> 1;
        for (int k = tid; k < (n >> 1); k += nt) {
            int grp = k / mh;
            int pos = k - grp * mh;
            int i0 = grp * m + pos;
            int i1 = i0 + mh;
            float ang = sign * 2.0f * PI_F * (float)pos / (float)m;
            float cw = cosf(ang);
            float sw = sinf(ang);
            float2 a = s[i0], b = s[i1];
            float2 t = make_float2(b.x * cw - b.y * sw, b.x * sw + b.y * cw);
            s[i0] = make_float2(a.x + t.x, a.y + t.y);
            s[i1] = make_float2(a.x - t.x, a.y - t.y);
        }
        __syncthreads();
    }
}

// ---------------------------------------------------------------------------
// 1) Cauchy + Woodbury. One block per channel h; threads stride over l.
// ---------------------------------------------------------------------------
__global__ void __launch_bounds__(256) cauchy_kernel(
    const float* __restrict__ Lre, const float* __restrict__ Lim,
    const float* __restrict__ Pre, const float* __restrict__ Pim,
    const float* __restrict__ Qre, const float* __restrict__ Qim,
    const float* __restrict__ Bre, const float* __restrict__ Bim,
    const float* __restrict__ Cre, const float* __restrict__ Cim,
    const float* __restrict__ step, float2* __restrict__ atRoots)
{
    __shared__ float2 lam_s[N_ST], qb_s[N_ST], qp_s[N_ST], cb_s[N_ST], cp_s[N_ST];
    int h = blockIdx.x;
    int tid = threadIdx.x;

    if (tid < N_ST) {
        int n = tid;
        float2 lam = make_float2(Lre[n], Lim[n]);
        float2 q = make_float2(Qre[n], Qim[n]);
        float2 b = make_float2(Bre[n], Bim[n]);
        float2 p = make_float2(Pre[n], Pim[n]);
        float2 cc = make_float2(Cre[h * N_ST + n], Cim[h * N_ST + n]);
        lam_s[n] = lam;
        qb_s[n] = cmul(q, b);
        qp_s[n] = cmul(q, p);
        cb_s[n] = cmul(cc, b);
        cp_s[n] = cmul(cc, p);
    }
    __syncthreads();

    float dt = fmaxf(step[0], 1e-6f);
    for (int l = tid; l < L_SEQ; l += 256) {
        float th = 2.0f * PI_F * (float)l / (float)L_SEQ;
        float2 Om = make_float2(cosf(th), -sinf(th));      // exp(-2*pi*i*l/L)
        float2 onePlus  = make_float2(1.0f + Om.x,  Om.y);
        float2 oneMinus = make_float2(1.0f - Om.x, -Om.y);
        float2 cfac = cscale(cinv(onePlus), 2.0f);
        float2 g    = cscale(cdiv(oneMinus, onePlus), 2.0f / dt);

        float2 k00 = make_float2(0.f, 0.f), k01 = make_float2(0.f, 0.f);
        float2 k10 = make_float2(0.f, 0.f), k11 = make_float2(0.f, 0.f);
#pragma unroll 4
        for (int n = 0; n < N_ST; ++n) {
            float2 iv = cinv(csub(g, lam_s[n]));
            k00 = cadd(k00, cmul(cb_s[n], iv));
            k01 = cadd(k01, cmul(cp_s[n], iv));
            k10 = cadd(k10, cmul(qb_s[n], iv));
            k11 = cadd(k11, cmul(qp_s[n], iv));
        }
        float2 corr = cdiv(k10, make_float2(1.0f + k11.x, k11.y));
        atRoots[(size_t)h * L_SEQ + l] = cmul(cfac, csub(k00, cmul(k01, corr)));
    }
}

// ---------------------------------------------------------------------------
// 2) Per channel: ifft(4096) -> real K -> zero-pad -> fft(8192) -> Kf
// ---------------------------------------------------------------------------
__global__ void __launch_bounds__(256) kgen_kernel(const float2* __restrict__ atRoots,
                                                   float2* __restrict__ Kf)
{
    __shared__ float2 s[NFFT];  // 64 KB of the 320 KB/WGP LDS
    int h = blockIdx.x, tid = threadIdx.x;
    for (int l = tid; l < L_SEQ; l += 256) s[l] = atRoots[(size_t)h * L_SEQ + l];
    __syncthreads();
    block_fft(s, L_SEQ, 12, +1.0f, tid, 256);              // ifft (unscaled)
    for (int l = tid; l < L_SEQ; l += 256) {
        float kr = s[l].x * (1.0f / (float)L_SEQ);
        s[l] = make_float2(kr, 0.0f);
    }
    for (int l = L_SEQ + tid; l < NFFT; l += 256) s[l] = make_float2(0.f, 0.f);
    __syncthreads();
    block_fft(s, NFFT, 13, -1.0f, tid, 256);               // forward fft
    for (int l = tid; l < NFFT; l += 256) Kf[(size_t)h * NFFT + l] = s[l];
}

// ---------------------------------------------------------------------------
// 3) Tiled transpose: x (B,L,H) -> xT (B,H,L). Coalesced both sides.
// ---------------------------------------------------------------------------
__global__ void __launch_bounds__(256) xpose_kernel(const float* __restrict__ in,
                                                    float* __restrict__ out)
{
    __shared__ float tile[32][33];
    int b = blockIdx.z;
    int l0 = blockIdx.x * 32;
    int h0 = blockIdx.y * 32;
    int tx = threadIdx.x, ty = threadIdx.y;
    const float* src = in + ((size_t)b * L_SEQ + l0) * H_CH + h0;
#pragma unroll
    for (int r = 0; r < 32; r += 8)
        tile[ty + r][tx] = src[(size_t)(ty + r) * H_CH + tx];
    __syncthreads();
    float* dst = out + ((size_t)b * H_CH + h0) * L_SEQ + l0;
#pragma unroll
    for (int r = 0; r < 32; r += 8)
        dst[(size_t)(ty + r) * L_SEQ + tx] = tile[tx][ty + r];
}

// ---------------------------------------------------------------------------
// 4) Per (b,h): fft(xT row, zero-padded) * Kf -> ifft -> h1T row (contiguous)
// ---------------------------------------------------------------------------
__global__ void __launch_bounds__(256) conv_kernel(const float* __restrict__ xT,
                                                   const float2* __restrict__ Kf,
                                                   float* __restrict__ h1T)
{
    __shared__ float2 s[NFFT];
    int b = blockIdx.x >> 8;
    int h = blockIdx.x & 255;
    int tid = threadIdx.x;
    const float* xr = xT + ((size_t)b * H_CH + h) * L_SEQ;
    for (int l = tid; l < L_SEQ; l += 256) s[l] = make_float2(xr[l], 0.f);
    for (int l = L_SEQ + tid; l < NFFT; l += 256) s[l] = make_float2(0.f, 0.f);
    __syncthreads();
    block_fft(s, NFFT, 13, -1.0f, tid, 256);
    const float2* kf = Kf + (size_t)h * NFFT;
    for (int l = tid; l < NFFT; l += 256) s[l] = cmul(s[l], kf[l]);
    __syncthreads();
    block_fft(s, NFFT, 13, +1.0f, tid, 256);
    float* o = h1T + ((size_t)b * H_CH + h) * L_SEQ;
    for (int l = tid; l < L_SEQ; l += 256) o[l] = s[l].x * (1.0f / (float)NFFT);
}

// ---------------------------------------------------------------------------
// 5) Fused transpose-back + pointwise: y1h[(b*L+l), h] = f16(gelu(h1T + D*x))
// ---------------------------------------------------------------------------
__global__ void __launch_bounds__(256) packT_kernel(const float* __restrict__ h1T,
                                                    const float* __restrict__ x,
                                                    const float* __restrict__ D,
                                                    _Float16* __restrict__ y1h)
{
    __shared__ float tile[32][33];
    int b = blockIdx.z;
    int h0 = blockIdx.x * 32;
    int l0 = blockIdx.y * 32;
    int tx = threadIdx.x, ty = threadIdx.y;
    const float* src = h1T + ((size_t)b * H_CH + h0) * L_SEQ + l0;
#pragma unroll
    for (int r = 0; r < 32; r += 8)
        tile[ty + r][tx] = src[(size_t)(ty + r) * L_SEQ + tx];   // coalesced in l
    __syncthreads();
    float dv = D[0];
    const float* xs = x + ((size_t)b * L_SEQ + l0) * H_CH + h0;
    _Float16* dst = y1h + ((size_t)b * L_SEQ + l0) * H_CH + h0;
#pragma unroll
    for (int r = 0; r < 32; r += 8) {
        float v = tile[tx][ty + r] + dv * xs[(size_t)(ty + r) * H_CH + tx];
        dst[(size_t)(ty + r) * H_CH + tx] = (_Float16)gelu_exact(v);   // coalesced in h
    }
}

// 6) W_fc -> f16
__global__ void wconv_kernel(const float* __restrict__ W, _Float16* __restrict__ Wh, int total)
{
    int i = blockIdx.x * 256 + threadIdx.x;
    if (i >= total) return;
    Wh[i] = (_Float16)W[i];
}

// 7) zero the max accumulators every launch (determinism under graph replay)
__global__ void init_max_kernel(unsigned* __restrict__ p, int total)
{
    int i = blockIdx.x * 256 + threadIdx.x;
    if (i < total) p[i] = 0u;
}

// ---------------------------------------------------------------------------
// 8) FC GEMM via WMMA. Block = 8 waves x 128 rows, one 64-col W panel in LDS
//    (32 KB, staged with async-to-LDS). K-loop fully unrolled: double-buffered
//    A becomes pure SSA renaming (no movs, no WMMA->VALU hazard NOPs); per
//    K-step all 4 B fragments batch-load then 4 back-to-back WMMAs.
//    grid (M/128, N/64); EXEC all-1s in the WMMA region.
// ---------------------------------------------------------------------------
__global__ void __launch_bounds__(256) fc_gemm_kernel(const _Float16* __restrict__ A,
                                                      const _Float16* __restrict__ Wh,
                                                      const float* __restrict__ bfc,
                                                      const float* __restrict__ x,
                                                      float* __restrict__ y2)
{
    __shared__ __align__(32) _Float16 Bl[64 * H_CH];   // 32 KB panel: 64 cols x 256 K

    int wave = threadIdx.x >> 5;
    int lane = threadIdx.x & 31;
    int panel = blockIdx.y;                       // 64-column panel of N
    int m0 = blockIdx.x * 128 + wave * 16;        // 8 waves -> 128 rows
    bool hi = lane >= 16;
    int lrow = lane & 15;

    // ---- stage W panel (cols panel*64 .. +63, all 256 K) into LDS ----
    const _Float16* wsrc = Wh + (size_t)(panel * 64) * H_CH;   // 32 KB contiguous
#ifdef HAVE_ASYNC_LDS
#pragma unroll
    for (int i = 0; i < 8; ++i) {
        int idx = (i * 256 + (int)threadIdx.x) * 8;            // halves, 16B chunks
        __builtin_amdgcn_global_load_async_to_lds_b128(
            (global_v4i*)(wsrc + idx), (lds_v4i*)(&Bl[idx]), 0, 0);
    }
#ifdef HAVE_WAIT_ASYNC
    __builtin_amdgcn_s_wait_asynccnt(0);
#else
    asm volatile("s_wait_asynccnt 0x0" ::: "memory");
#endif
#else
    {
        const float4* s4 = (const float4*)wsrc;
        float4* d4 = (float4*)Bl;
        for (int i = threadIdx.x; i < (64 * H_CH) / 8; i += 256) d4[i] = s4[i];
    }
#endif
    __syncthreads();

    v8f acc[4];
#pragma unroll
    for (int t = 0; t < 4; ++t) acc[t] = (v8f){0.f, 0.f, 0.f, 0.f, 0.f, 0.f, 0.f, 0.f};

    const _Float16* arow = A + (size_t)(m0 + lrow) * H_CH;

    // A fragment for k0 = 0 (contiguous K order matching B: lanes 0-15 hold K
    // {0..15}, lanes 16-31 hold {16..31})
    v16h a;
    {
        const _Float16* ap = arow + (hi ? 16 : 0);
#pragma unroll
        for (int j = 0; j < 8; ++j) { a[j] = ap[j]; a[8 + j] = ap[8 + j]; }
    }

#pragma unroll
    for (int k0 = 0; k0 < H_CH; k0 += 32) {
        // double-buffer: issue next A-fragment loads before this step's WMMAs
        int kn = (k0 + 32 < H_CH) ? (k0 + 32) : k0;
        const _Float16* ap = arow + kn + (hi ? 16 : 0);
        v16h an;
#pragma unroll
        for (int j = 0; j < 8; ++j) { an[j] = ap[j]; an[8 + j] = ap[8 + j]; }

        // Phase 1: batch all 4 B fragments into distinct registers
        v16h bv[4];
#pragma unroll
        for (int t = 0; t < 4; ++t) {
            const v16h* bp = (const v16h*)&Bl[(size_t)(t * 16 + lrow) * H_CH + k0 + (hi ? 16 : 0)];
            bv[t] = *bp;
        }
        // Phase 2: 4 back-to-back WMMAs on the matrix pipe
#pragma unroll
        for (int t = 0; t < 4; ++t) {
            acc[t] = __builtin_amdgcn_wmma_f32_16x16x32_f16(
                false, a, false, bv[t], (short)0, acc[t], false, false);
        }
        a = an;
    }

    // Epilogue: D layout — vgpr r holds (m = m0 + r + (hi?8:0), n = t*16 + lrow)
#pragma unroll
    for (int t = 0; t < 4; ++t) {
        int n = panel * 64 + t * 16 + lrow;
        float bb = bfc[n];
#pragma unroll
        for (int r = 0; r < 8; ++r) {
            int m = m0 + r + (hi ? 8 : 0);
            size_t idx = (size_t)m * H_CH + n;
            float v = gelu_exact(acc[t][r] + bb);
            y2[idx] = v + x[idx];
        }
    }
}

// ---------------------------------------------------------------------------
// 9) LayerNorm over H, write y, accumulate per-(b,h) max|y| and max|x|
// ---------------------------------------------------------------------------
__global__ void __launch_bounds__(256) ln_kernel(const float* __restrict__ y2,
                                                 const float* __restrict__ x,
                                                 const float* __restrict__ lng,
                                                 const float* __restrict__ lnb,
                                                 float* __restrict__ out,
                                                 unsigned* __restrict__ maxy,
                                                 unsigned* __restrict__ maxx)
{
    __shared__ float red[256];
    __shared__ float s_mean, s_var;
    int m = blockIdx.x, h = threadIdx.x;
    size_t idx = (size_t)m * H_CH + h;
    float v = y2[idx];

    red[h] = v;
    __syncthreads();
    for (int s2 = 128; s2 > 0; s2 >>= 1) {
        if (h < s2) red[h] += red[h + s2];
        __syncthreads();
    }
    if (h == 0) s_mean = red[0] * (1.0f / (float)H_CH);
    __syncthreads();

    float d = v - s_mean;
    red[h] = d * d;
    __syncthreads();
    for (int s2 = 128; s2 > 0; s2 >>= 1) {
        if (h < s2) red[h] += red[h + s2];
        __syncthreads();
    }
    if (h == 0) s_var = red[0] * (1.0f / (float)H_CH);
    __syncthreads();

    float y = d * rsqrtf(s_var + 1e-5f) * lng[h] + lnb[h];
    out[idx] = y;

    int b = m >> 12;  // m / L_SEQ
    atomicMax(&maxy[b * H_CH + h], __float_as_uint(fabsf(y)));    // |v|>=0: uint order == float order
    atomicMax(&maxx[b * H_CH + h], __float_as_uint(fabsf(x[idx])));
}

// 10) h_ratio
__global__ void ratio_kernel(const unsigned* __restrict__ maxy, const unsigned* __restrict__ maxx,
                             float* __restrict__ outr)
{
    int i = blockIdx.x * 256 + threadIdx.x;
    if (i >= B_SZ * H_CH) return;
    float my = __uint_as_float(maxy[i]);
    float mx = __uint_as_float(maxx[i]);
    outr[i] = my / (mx + 1e-6f);
}

// ---------------------------------------------------------------------------
extern "C" void kernel_launch(void* const* d_in, const int* in_sizes, int n_in,
                              void* d_out, int out_size, void* d_ws, size_t ws_size,
                              hipStream_t stream)
{
    (void)in_sizes; (void)n_in; (void)out_size; (void)ws_size;

    const float* x    = (const float*)d_in[0];
    const float* Lre  = (const float*)d_in[1];
    const float* Lim  = (const float*)d_in[2];
    const float* Pre  = (const float*)d_in[3];
    const float* Pim  = (const float*)d_in[4];
    const float* Qre  = (const float*)d_in[5];
    const float* Qim  = (const float*)d_in[6];
    const float* Bre  = (const float*)d_in[7];
    const float* Bim  = (const float*)d_in[8];
    const float* Cre  = (const float*)d_in[9];
    const float* Cim  = (const float*)d_in[10];
    const float* step = (const float*)d_in[11];
    const float* D    = (const float*)d_in[12];
    const float* Wfc  = (const float*)d_in[13];
    const float* bfc  = (const float*)d_in[14];
    const float* lng  = (const float*)d_in[15];
    const float* lnb  = (const float*)d_in[16];

    float* outp = (float*)d_out;
    const size_t total = (size_t)B_SZ * L_SEQ * H_CH;   // 8,388,608

    // Workspace layout (~136 MB)
    char* ws = (char*)d_ws;
    size_t off = 0;
    auto carve = [&](size_t bytes) -> char* {
        char* p = ws + off;
        off += (bytes + 255) & ~(size_t)255;
        return p;
    };
    float2*   atR  = (float2*)  carve((size_t)H_CH * L_SEQ * sizeof(float2));  //  8 MB
    float2*   Kf   = (float2*)  carve((size_t)H_CH * NFFT  * sizeof(float2));  // 16 MB
    float*    xT   = (float*)   carve(total * sizeof(float));                  // 32 MB
    float*    h1T  = (float*)   carve(total * sizeof(float));                  // 32 MB
    _Float16* y1h  = (_Float16*)carve(total * sizeof(_Float16));               // 16 MB
    _Float16* Wh   = (_Float16*)carve((size_t)H_CH * H_CH * sizeof(_Float16)); // 128 KB
    float*    y2   = (float*)   carve(total * sizeof(float));                  // 32 MB
    unsigned* maxy = (unsigned*)carve((size_t)B_SZ * H_CH * sizeof(unsigned));
    unsigned* maxx = (unsigned*)carve((size_t)B_SZ * H_CH * sizeof(unsigned));

    // 1) kernel generation in the frequency domain (coalesced per-channel rows)
    cauchy_kernel<<<H_CH, 256, 0, stream>>>(Lre, Lim, Pre, Pim, Qre, Qim,
                                            Bre, Bim, Cre, Cim, step, atR);
    // 2) ifft + zero-padded fft per channel
    kgen_kernel<<<H_CH, 256, 0, stream>>>(atR, Kf);
    // 3) transpose x for contiguous per-(b,h) rows
    xpose_kernel<<<dim3(L_SEQ / 32, H_CH / 32, B_SZ), dim3(32, 8), 0, stream>>>(x, xT);
    // 4) FFT convolution per (b, h)
    conv_kernel<<<B_SZ * H_CH, 256, 0, stream>>>(xT, Kf, h1T);
    // 5) fused gelu + transpose back -> f16 A matrix
    packT_kernel<<<dim3(H_CH / 32, L_SEQ / 32, B_SZ), dim3(32, 8), 0, stream>>>(h1T, x, D, y1h);
    // 6) W -> f16
    wconv_kernel<<<(H_CH * H_CH) / 256, 256, 0, stream>>>(Wfc, Wh, H_CH * H_CH);
    // 7) zero max accumulators (maxy and maxx are contiguous)
    init_max_kernel<<<(2 * B_SZ * H_CH + 255) / 256, 256, 0, stream>>>(maxy, 2 * B_SZ * H_CH);
    // 8) FC layer on the WMMA pipe (LDS-resident W panel)
    fc_gemm_kernel<<<dim3((B_SZ * L_SEQ) / 128, H_CH / 64), 256, 0, stream>>>(y1h, Wh, bfc, x, y2);
    // 9) LayerNorm + maxes
    ln_kernel<<<B_SZ * L_SEQ, 256, 0, stream>>>(y2, x, lng, lnb, outp, maxy, maxx);
    // 10) h_ratio tail
    ratio_kernel<<<(B_SZ * H_CH + 255) / 256, 256, 0, stream>>>(maxy, maxx, outp + total);
}
